// TrajectoryDecoder_65652870087173
// MI455X (gfx1250) — compile-verified, matrix operand order checked by reading, not run on hardware
//
#include <hip/hip_runtime.h>
#include <hip/hip_bf16.h>

// ---------------------------------------------------------------------------
// Problem constants (reference: B,A,H,O,T = 64,256,512,2,32)
// ---------------------------------------------------------------------------
#define Bc 64
#define Ac 256
#define Hc 512
#define Oc 2
#define Tc 32
constexpr int NR = Bc * Ac;   // 16384 rows
constexpr int G3 = 3 * Hc;    // 1536 gate columns
constexpr int HH = Hc / 2;    // 256 MLP hidden

typedef __attribute__((ext_vector_type(16))) __bf16 bf16x16;
typedef __attribute__((ext_vector_type(8)))  float  f32x8;

static __device__ __forceinline__ unsigned short f32_to_bf16_rne(float f) {
  unsigned int u = __float_as_uint(f);
  u += 0x7FFFu + ((u >> 16) & 1u);   // round-to-nearest-even
  return (unsigned short)(u >> 16);
}
static __device__ __forceinline__ float sigmoidf_(float v) {
  return 1.0f / (1.0f + __expf(-v));
}

// ---------------------------------------------------------------------------
// fp32 -> bf16 conversion (weights / activation mirrors)
// ---------------------------------------------------------------------------
__global__ void cvt_f32_bf16_k(const float* __restrict__ src,
                               unsigned short* __restrict__ dst, size_t n) {
  size_t i = (size_t)blockIdx.x * blockDim.x + threadIdx.x;
  if (i < n) dst[i] = f32_to_bf16_rne(src[i]);
}

// ---------------------------------------------------------------------------
// Init hidden state (replicates the reference's permuted init:
// h_init row r = social_features[b=r%B, a=r/B]) and x0 = last_position rows.
// ---------------------------------------------------------------------------
__global__ void init_state_k(const float* __restrict__ social,
                             const float* __restrict__ lastp,
                             float* __restrict__ h0f, float* __restrict__ h1f,
                             unsigned short* __restrict__ h0b,
                             unsigned short* __restrict__ h1b,
                             float* __restrict__ x) {
  int r = blockIdx.x;                 // GRU row 0..NR-1
  int a = r / Bc, b = r % Bc;         // r = a*B + b
  const float* srow = social + ((size_t)b * Ac + a) * Hc;
  for (int j = threadIdx.x; j < Hc; j += blockDim.x) {
    float v = srow[j];
    size_t idx = (size_t)r * Hc + j;
    h0f[idx] = v; h1f[idx] = v;
    unsigned short bv = f32_to_bf16_rne(v);
    h0b[idx] = bv; h1b[idx] = bv;
  }
  if (threadIdx.x < Oc)               // x rows are ordered b*A + a == r
    x[(size_t)r * Oc + threadIdx.x] = lastp[(size_t)r * Oc + threadIdx.x];
}

// ---------------------------------------------------------------------------
// WMMA bf16 GEMM:  C[M,N](f32) = A[M,K](bf16) @ B[N,K](bf16)^T + bias[N]
// Block: 256 threads (8 waves) -> block tile 128(M) x 256(N).
// Wave tile 64x64 = 4x4 WMMA tiles: 16 WMMAs per 8 fragment loads per k-step
// (~32 FLOP per operand byte -> operand-bandwidth friendly).
// Grid: (N/256, M/128). Optional ReLU epilogue.
// ---------------------------------------------------------------------------
__global__ __launch_bounds__(256)
void gemm_bf16_nt(const unsigned short* __restrict__ A,
                  const unsigned short* __restrict__ Bw,
                  const float* __restrict__ bias,
                  float* __restrict__ C,
                  int M, int N, int K, int relu) {
  const int tid  = threadIdx.x;
  const int wave = tid >> 5;
  const int lane = tid & 31;
  const int lo = lane & 15, hi = lane >> 4;

  const int bm = blockIdx.y * 128;
  const int bn = blockIdx.x * 256;
  const int wm = (wave & 1) * 64;   // 2 waves over M -> 4 row-tiles each
  const int wn = (wave >> 1) * 64;  // 4 waves over N -> 4 col-tiles each

  f32x8 acc[4][4];
#pragma unroll
  for (int i = 0; i < 4; ++i)
#pragma unroll
    for (int j = 0; j < 4; ++j) acc[i][j] = f32x8{};

  // Contiguous k order k = 16*hi + s (same in A and B, not the ISA table order): lane = row (lo), lanes>=16 hold K+16..31 (hi).
  const unsigned short* Aptr = A  + (size_t)(bm + wm + lo) * K + hi * 16;
  const unsigned short* Bptr = Bw + (size_t)(bn + wn + lo) * K + hi * 16;

#pragma unroll 2
  for (int k0 = 0; k0 < K; k0 += 32) {
    bf16x16 afrag[4], bfrag[4];
#pragma unroll
    for (int i = 0; i < 4; ++i)
      afrag[i] = *(const bf16x16*)(Aptr + (size_t)(i * 16) * K + k0);
#pragma unroll
    for (int j = 0; j < 4; ++j)
      bfrag[j] = *(const bf16x16*)(Bptr + (size_t)(j * 16) * K + k0);
#pragma unroll
    for (int i = 0; i < 4; ++i)
#pragma unroll
      for (int j = 0; j < 4; ++j)
        acc[i][j] = __builtin_amdgcn_wmma_f32_16x16x32_bf16(
            false, afrag[i], false, bfrag[j], (short)0, acc[i][j],
            false, false);
  }

  // Epilogue: C/D layout — VGPR v: row = tile_m + v + 8*hi, col = tile_n + lo.
#pragma unroll
  for (int i = 0; i < 4; ++i) {
    int mrow = bm + wm + i * 16 + hi * 8;
#pragma unroll
    for (int j = 0; j < 4; ++j) {
      int col = bn + wn + j * 16 + lo;
      float bv = bias ? bias[col] : 0.0f;
#pragma unroll
      for (int v = 0; v < 8; ++v) {
        float val = acc[i][j][v] + bv;
        if (relu) val = fmaxf(val, 0.0f);
        C[(size_t)(mrow + v) * N + col] = val;
      }
    }
  }
}

// ---------------------------------------------------------------------------
// GRU cell layer 0: gi computed inline from x (K=O=2), gh from GEMM (incl b_hh0).
// ---------------------------------------------------------------------------
__global__ __launch_bounds__(256)
void gru_cell0_k(const float* __restrict__ x,
                 const float* __restrict__ Wih0, const float* __restrict__ bih0,
                 const float* __restrict__ gh,
                 float* __restrict__ hf, unsigned short* __restrict__ hb) {
  size_t idx = (size_t)blockIdx.x * blockDim.x + threadIdx.x; // over NR*Hc
  int r = (int)(idx >> 9);          // /Hc
  int j = (int)(idx & (Hc - 1));
  float x0 = x[(size_t)r * Oc], x1 = x[(size_t)r * Oc + 1];
  float ir = x0 * Wih0[(size_t)j * 2]            + x1 * Wih0[(size_t)j * 2 + 1]            + bih0[j];
  float iz = x0 * Wih0[(size_t)(j + Hc) * 2]     + x1 * Wih0[(size_t)(j + Hc) * 2 + 1]     + bih0[j + Hc];
  float in_= x0 * Wih0[(size_t)(j + 2 * Hc) * 2] + x1 * Wih0[(size_t)(j + 2 * Hc) * 2 + 1] + bih0[j + 2 * Hc];
  const float* ghr = gh + (size_t)r * G3;
  float rg = sigmoidf_(ir + ghr[j]);
  float zg = sigmoidf_(iz + ghr[j + Hc]);
  float ng = tanhf(in_ + rg * ghr[j + 2 * Hc]);
  float h  = hf[idx];
  float hn = (1.0f - zg) * ng + zg * h;
  hf[idx] = hn;
  hb[idx] = f32_to_bf16_rne(hn);
}

// ---------------------------------------------------------------------------
// GRU cell layer 1: gi and gh both come from GEMMs (biases already added).
// ---------------------------------------------------------------------------
__global__ __launch_bounds__(256)
void gru_cell1_k(const float* __restrict__ gi, const float* __restrict__ gh,
                 float* __restrict__ hf, unsigned short* __restrict__ hb) {
  size_t idx = (size_t)blockIdx.x * blockDim.x + threadIdx.x;
  int r = (int)(idx >> 9);
  int j = (int)(idx & (Hc - 1));
  const float* gir = gi + (size_t)r * G3;
  const float* ghr = gh + (size_t)r * G3;
  float rg = sigmoidf_(gir[j] + ghr[j]);
  float zg = sigmoidf_(gir[j + Hc] + ghr[j + Hc]);
  float ng = tanhf(gir[j + 2 * Hc] + rg * ghr[j + 2 * Hc]);
  float h  = hf[idx];
  float hn = (1.0f - zg) * ng + zg * h;
  hf[idx] = hn;
  hb[idx] = f32_to_bf16_rne(hn);
}

// ---------------------------------------------------------------------------
// Output head: m (NR x HH, already ReLU'ed) @ W2^T + b2 -> out[b,t,a,:], x_next.
// ---------------------------------------------------------------------------
__global__ __launch_bounds__(256)
void pred_k(const float* __restrict__ m, const float* __restrict__ W2,
            const float* __restrict__ b2, float* __restrict__ out,
            float* __restrict__ x, int t) {
  int r = blockIdx.x * blockDim.x + threadIdx.x;
  if (r >= NR) return;
  const float* mr = m + (size_t)r * HH;
  float p0 = b2[0], p1 = b2[1];
  for (int k = 0; k < HH; ++k) {
    float mv = mr[k];
    p0 += mv * W2[k];
    p1 += mv * W2[HH + k];
  }
  int b = r / Ac, a = r % Ac;
  size_t o = (((size_t)b * Tc + t) * Ac + a) * Oc;
  out[o] = p0; out[o + 1] = p1;
  x[(size_t)r * Oc] = p0; x[(size_t)r * Oc + 1] = p1;   // autoregressive input
}

// ---------------------------------------------------------------------------
// Host-side orchestration (graph-capture safe: only launches on `stream`).
// ---------------------------------------------------------------------------
extern "C" void kernel_launch(void* const* d_in, const int* in_sizes, int n_in,
                              void* d_out, int out_size, void* d_ws, size_t ws_size,
                              hipStream_t stream) {
  const float* social = (const float*)d_in[0];
  const float* lastp  = (const float*)d_in[1];
  const float* W_ih0  = (const float*)d_in[2];
  const float* W_hh0  = (const float*)d_in[3];
  const float* b_ih0  = (const float*)d_in[4];
  const float* b_hh0  = (const float*)d_in[5];
  const float* W_ih1  = (const float*)d_in[6];
  const float* W_hh1  = (const float*)d_in[7];
  const float* b_ih1  = (const float*)d_in[8];
  const float* b_hh1  = (const float*)d_in[9];
  const float* W1     = (const float*)d_in[10];
  const float* b1     = (const float*)d_in[11];
  const float* W2     = (const float*)d_in[12];
  const float* b2     = (const float*)d_in[13];
  float* out = (float*)d_out;

  // Workspace carve-out (256B aligned slabs).
  char* ws = (char*)d_ws;
  size_t off = 0;
  auto take = [&](size_t bytes) -> char* {
    char* p = ws + off;
    off = (off + bytes + 255) & ~(size_t)255;
    return p;
  };
  unsigned short* wb_hh0 = (unsigned short*)take((size_t)G3 * Hc * 2);
  unsigned short* wb_ih1 = (unsigned short*)take((size_t)G3 * Hc * 2);
  unsigned short* wb_hh1 = (unsigned short*)take((size_t)G3 * Hc * 2);
  unsigned short* wb_w1  = (unsigned short*)take((size_t)HH * Hc * 2);
  float*          h0f    = (float*)take((size_t)NR * Hc * 4);
  float*          h1f    = (float*)take((size_t)NR * Hc * 4);
  unsigned short* h0b    = (unsigned short*)take((size_t)NR * Hc * 2);
  unsigned short* h1b    = (unsigned short*)take((size_t)NR * Hc * 2);
  float*          g1     = (float*)take((size_t)NR * G3 * 4);
  float*          g2     = (float*)take((size_t)NR * G3 * 4);
  float*          mbuf   = (float*)take((size_t)NR * HH * 4);
  float*          xbuf   = (float*)take((size_t)NR * Oc * 4);
  (void)ws_size; (void)in_sizes; (void)n_in; (void)out_size;

  // 1) Weights -> bf16 (every call; deterministic).
  auto cvt = [&](const float* s, unsigned short* d, size_t n) {
    cvt_f32_bf16_k<<<(unsigned)((n + 255) / 256), 256, 0, stream>>>(s, d, n);
  };
  cvt(W_hh0, wb_hh0, (size_t)G3 * Hc);
  cvt(W_ih1, wb_ih1, (size_t)G3 * Hc);
  cvt(W_hh1, wb_hh1, (size_t)G3 * Hc);
  cvt(W1,    wb_w1,  (size_t)HH * Hc);

  // 2) Hidden/x init (permuted mapping baked in).
  init_state_k<<<NR, 128, 0, stream>>>(social, lastp, h0f, h1f, h0b, h1b, xbuf);

  // 3) T sequential decode steps.
  dim3 gemm_grid_g(G3 / 256, NR / 128);   // (6, 128)
  dim3 gemm_grid_m(HH / 256, NR / 128);   // (1, 128)
  const unsigned cellBlocks = (unsigned)(((size_t)NR * Hc) / 256);
  for (int t = 0; t < Tc; ++t) {
    // layer 0: gh0 = h0 @ W_hh0^T + b_hh0
    gemm_bf16_nt<<<gemm_grid_g, 256, 0, stream>>>(h0b, wb_hh0, b_hh0, g1,
                                                  NR, G3, Hc, 0);
    gru_cell0_k<<<cellBlocks, 256, 0, stream>>>(xbuf, W_ih0, b_ih0, g1, h0f, h0b);
    // layer 1: gi1 = h0n @ W_ih1^T + b_ih1 ; gh1 = h1 @ W_hh1^T + b_hh1
    gemm_bf16_nt<<<gemm_grid_g, 256, 0, stream>>>(h0b, wb_ih1, b_ih1, g1,
                                                  NR, G3, Hc, 0);
    gemm_bf16_nt<<<gemm_grid_g, 256, 0, stream>>>(h1b, wb_hh1, b_hh1, g2,
                                                  NR, G3, Hc, 0);
    gru_cell1_k<<<cellBlocks, 256, 0, stream>>>(g1, g2, h1f, h1b);
    // head: m = relu(h1n @ W1^T + b1); pred = m @ W2^T + b2
    gemm_bf16_nt<<<gemm_grid_m, 256, 0, stream>>>(h1b, wb_w1, b1, mbuf,
                                                  NR, HH, Hc, 1);
    pred_k<<<NR / 256, 256, 0, stream>>>(mbuf, W2, b2, out, xbuf, t);
  }
}
